// TimeStepLayer_3624952398586
// MI455X (gfx1250) — matrix operand order repaired and checked by reading, not compiled, not run on hardware
//
#include <hip/hip_runtime.h>

#define UNITS  1024
#define TSTEPS 512
#define NDIMS  7
#define DTOT   1031          // 7 + 1024
#define HH     16
#define K3     48
#define NCHUNK 33            // ceil(1031/32)
#define NWG    128
#define TPB    512           // 16 waves
#define PAIRS_PER_WG 8       // 2 waves per unit -> 8 units per WG, 128*8 = 1024 units

typedef __bf16 v16bf __attribute__((ext_vector_type(16)));
typedef float  v8f   __attribute__((ext_vector_type(8)));

__device__ __forceinline__ unsigned short f2bf(float f) {
  unsigned u = __float_as_uint(f);
  unsigned r = u + 0x7FFFu + ((u >> 16) & 1u);   // round-to-nearest-even
  return (unsigned short)(r >> 16);
}
__device__ __forceinline__ float sigmoidf_(float x) { return 1.0f / (1.0f + __expf(-x)); }

// ---------------------------------------------------------------------------
// Zero the per-step global-barrier counters (deterministic init every call).
__global__ void init_ctr_kernel(unsigned* __restrict__ ctr) {
  ctr[threadIdx.x] = 0u;
}

// ---------------------------------------------------------------------------
// One-time (per call) conversion of W1 (f32, [u,d,k]) into bf16 B-tiles laid
// out exactly in WMMA B-register order. One thread per (u, tile*33+c, lane):
// reads 16 strided f32 (coalesced across lanes: 16 consecutive k per row),
// writes one contiguous 32B block (wave writes a full 1KB tile contiguously).
__global__ __launch_bounds__(256)
void convert_w1_kernel(const float* __restrict__ W1, unsigned short* __restrict__ wt) {
  int widx = blockIdx.x * 256 + threadIdx.x;     // 0 .. 1024*99*32-1
  int lane = widx & 31;
  int tc   = (widx >> 5) % 99;                   // tile*33 + c
  int u    = (widx >> 5) / 99;
  int tile = tc / 33;
  int c    = tc % 33;
  int koff = (lane >> 4) << 4;                   // lanes 16..31 hold K=16..31
  int k    = tile * 16 + (lane & 15);
  const float* src = W1 + (size_t)u * DTOT * K3 + k;
  unsigned dw[8];
#pragma unroll
  for (int v = 0; v < 8; ++v) {
    int d0 = c * 32 + koff + 2 * v;
    float f0 = (d0     < DTOT) ? src[(size_t)d0 * K3]       : 0.0f;
    float f1 = (d0 + 1 < DTOT) ? src[(size_t)(d0 + 1) * K3] : 0.0f;
    dw[v] = (unsigned)f2bf(f0) | ((unsigned)f2bf(f1) << 16);
  }
  uint4* dst = (uint4*)((char*)wt + ((size_t)(u * 99 + tc) * 1024 + (size_t)lane * 32));
  dst[0] = make_uint4(dw[0], dw[1], dw[2], dw[3]);
  dst[1] = make_uint4(dw[4], dw[5], dw[6], dw[7]);
}

// ---------------------------------------------------------------------------
// Persistent scan kernel: 512 sequential steps with a device-wide barrier per
// step; d_out row (t-1) is the recurrent state. Two waves split one unit's
// 33 K-chunks; 3 bf16 WMMAs per chunk (one per 16-wide k-tile of the 48).
__global__ __launch_bounds__(TPB)
void scan_kernel(const float* __restrict__ x,   const float* __restrict__ state0,
                 const float* __restrict__ b1i, const float* __restrict__ b1r,
                 const float* __restrict__ W2,  const float* __restrict__ b2i,
                 const float* __restrict__ b2r, const float* __restrict__ Wd,
                 const float* __restrict__ bd,
                 const unsigned short* __restrict__ wt,
                 unsigned* __restrict__ ctr, float* __restrict__ out) {
  __shared__ __align__(32) unsigned short ABuf[NCHUNK][2][16]; // pre-permuted A rows
  __shared__ float PBuf[PAIRS_PER_WG][2][3][16];               // per-wave partial mx1

  const int tid  = threadIdx.x;
  const int lane = tid & 31;
  const int wave = tid >> 5;
  const int q    = wave >> 1;            // unit-pair id within WG
  const int u    = blockIdx.x * PAIRS_PER_WG + q;
  const int n15  = lane & 15;
  const int half = lane >> 4;
  // Wave-uniform scalar: which chunk parity this wave handles. readfirstlane
  // forces it into an SGPR so the hot loop is a uniform (scalar) loop.
  const int par_s = __builtin_amdgcn_readfirstlane(wave & 1);

  const unsigned short* wtu = wt + (size_t)u * (99u * 512u);   // elements (bf16)
  const float*          w2u = W2 + (size_t)u * (HH * K3);

  // Hoist per-unit, time-invariant scalars out of the scan loop.
  const int ub = u * K3;
  const float c_b1z  = b1i[ub + n15]      + b1r[ub + n15];
  const float c_b1r  = b1i[ub + 16 + n15] + b1r[ub + 16 + n15];
  const float c_b1hi = b1i[ub + 32 + n15];
  const float c_b1hr = b1r[ub + 32 + n15];
  const float c_b2A  = b2i[ub + lane];
  const float c_b2B  = b2i[ub + 32 + n15];
  const float c_b2rz = b2r[ub + n15];
  const float c_b2rr = b2r[ub + 16 + n15];
  const float c_b2rh = b2r[ub + 32 + n15];
  const float c_wd   = Wd[u * HH + n15];
  const float c_bd   = bd[u];

  for (int t = 0; t < TSTEPS; ++t) {
    const float* prev = (t == 0) ? state0 : (out + (size_t)(t - 1) * UNITS);

    // Stage xin chunks into LDS, permuted into this kernel's contiguous-k A-register order (same as wt):
    // half 0 holds K=0..15 ; half 1 holds K=16..31.
    // 33*32 = 1056 = 2*TPB + 32 entries -> two full passes + small remainder.
#pragma unroll
    for (int pass = 0; pass < 2; ++pass) {
      int idx = tid + pass * TPB;
      int c = idx >> 5, j = idx & 31;
      int d = c * 32 + j;
      float v = (d < NDIMS) ? x[t * NDIMS + d]
                            : ((d < DTOT) ? prev[d - NDIMS] : 0.0f);
      int hf  = j >> 4;
      int pos = j & 15;
      ABuf[c][hf][pos] = f2bf(v);
    }
    if (tid < 32) {
      int idx = tid + 2 * TPB;            // c == 32 remainder chunk
      int j = idx & 31;
      int d = 1024 + j;
      float v = (d < DTOT) ? prev[d - NDIMS] : 0.0f;
      int hf  = j >> 4;
      int pos = j & 15;
      ABuf[32][hf][pos] = f2bf(v);
    }
    __syncthreads();

    v8f a0 = {}, a1 = {}, a2 = {};
    // Uniform 16-iteration loop: c = par_s, par_s+2, ..., par_s+30.
#pragma unroll 2
    for (int i = 0; i < 16; ++i) {
      int c = par_s + 2 * i;
      v16bf A  = *(const v16bf*)(&ABuf[c][half][0]);
      const unsigned short* bb = wtu + (size_t)c * 512 + (size_t)lane * 16;
      v16bf B0 = *(const v16bf*)(bb);
      v16bf B1 = *(const v16bf*)(bb + 33 * 512);
      v16bf B2 = *(const v16bf*)(bb + 66 * 512);
      a0 = __builtin_amdgcn_wmma_f32_16x16x32_bf16(false, A, false, B0, (short)0, a0, false, false);
      a1 = __builtin_amdgcn_wmma_f32_16x16x32_bf16(false, A, false, B1, (short)0, a1, false, false);
      a2 = __builtin_amdgcn_wmma_f32_16x16x32_bf16(false, A, false, B2, (short)0, a2, false, false);
    }
    if (par_s == 0) {                     // tail chunk c == 32 (scalar branch)
      v16bf A  = *(const v16bf*)(&ABuf[32][half][0]);
      const unsigned short* bb = wtu + (size_t)32 * 512 + (size_t)lane * 16;
      v16bf B0 = *(const v16bf*)(bb);
      v16bf B1 = *(const v16bf*)(bb + 33 * 512);
      v16bf B2 = *(const v16bf*)(bb + 66 * 512);
      a0 = __builtin_amdgcn_wmma_f32_16x16x32_bf16(false, A, false, B0, (short)0, a0, false, false);
      a1 = __builtin_amdgcn_wmma_f32_16x16x32_bf16(false, A, false, B1, (short)0, a1, false, false);
      a2 = __builtin_amdgcn_wmma_f32_16x16x32_bf16(false, A, false, B2, (short)0, a2, false, false);
    }
    // All D rows are identical (A rows broadcast); row M=0 lives in acc[0].
    if (lane < 16) {
      PBuf[q][par_s][0][lane] = a0[0];
      PBuf[q][par_s][1][lane] = a1[0];
      PBuf[q][par_s][2][lane] = a2[0];
    }
    __syncthreads();

    if (par_s == 0) {  // even wave of each pair finishes the unit
      float m0 = PBuf[q][0][0][n15] + PBuf[q][1][0][n15];
      float m1 = PBuf[q][0][1][n15] + PBuf[q][1][1][n15];
      float m2 = PBuf[q][0][2][n15] + PBuf[q][1][2][n15];
      // GRU1 with h=0: rec = b1r, h1 = (1-z)*hh
      float z   = sigmoidf_(m0 + c_b1z);
      float r   = sigmoidf_(m1 + c_b1r);
      float hh1 = tanhf    (m2 + c_b1hi + r * c_b1hr);
      float h1  = (1.0f - z) * hh1;
      // mx2 = h1 @ W2 + b2i : lane holds k=lane (0..31) and k=32+n15
      float mA = c_b2A;
      float mB = c_b2B;
#pragma unroll
      for (int j = 0; j < HH; ++j) {
        float hv = __shfl(h1, j, 32);
        mA += hv * w2u[j * K3 + lane];
        mB += hv * w2u[j * K3 + 32 + n15];
      }
      float mz = __shfl(mA, n15, 32);        // k = n
      float mr = __shfl(mA, n15 + 16, 32);   // k = 16+n
      float mh = __shfl(mB, n15, 32);        // k = 32+n
      float z2  = sigmoidf_(mz + c_b2rz);
      float r2  = sigmoidf_(mr + c_b2rr);
      float hh2 = tanhf    (mh + r2 * c_b2rh);
      float h2  = (1.0f - z2) * hh2;
      // out = relu(h2 . Wd + bd)
      float p = (lane < 16) ? h2 * c_wd : 0.0f;
      p += __shfl_xor(p, 1, 32);
      p += __shfl_xor(p, 2, 32);
      p += __shfl_xor(p, 4, 32);
      p += __shfl_xor(p, 8, 32);
      if (lane == 0) out[(size_t)t * UNITS + u] = fmaxf(p + c_bd, 0.0f);
    }
    __syncthreads();
    // Device-wide barrier: one release-arrival per block, acquire-spin.
    if (tid == 0) {
      __hip_atomic_fetch_add(&ctr[t], 1u, __ATOMIC_RELEASE, __HIP_MEMORY_SCOPE_AGENT);
      while (__hip_atomic_load(&ctr[t], __ATOMIC_ACQUIRE, __HIP_MEMORY_SCOPE_AGENT) != (unsigned)NWG) {
        __builtin_amdgcn_s_sleep(1);
      }
    }
    __syncthreads();
  }
}

// ---------------------------------------------------------------------------
extern "C" void kernel_launch(void* const* d_in, const int* in_sizes, int n_in,
                              void* d_out, int out_size, void* d_ws, size_t ws_size,
                              hipStream_t stream) {
  (void)in_sizes; (void)n_in; (void)out_size; (void)ws_size;
  const float* x   = (const float*)d_in[0];
  const float* s0  = (const float*)d_in[1];
  const float* W1  = (const float*)d_in[2];
  /* d_in[3] = U1: dead (h==0 in reference) */
  const float* b1i = (const float*)d_in[4];
  const float* b1r = (const float*)d_in[5];
  const float* W2  = (const float*)d_in[6];
  /* d_in[7] = U2: dead */
  const float* b2i = (const float*)d_in[8];
  const float* b2r = (const float*)d_in[9];
  const float* Wd  = (const float*)d_in[10];
  const float* bd  = (const float*)d_in[11];

  unsigned*       ctr = (unsigned*)d_ws;                          // 512 * 4B
  unsigned short* wt  = (unsigned short*)((char*)d_ws + 4096);    // ~99 MiB bf16 tiles
  float*          out = (float*)d_out;

  init_ctr_kernel<<<1, TSTEPS, 0, stream>>>(ctr);
  // 1024 * 99 * 32 threads, 256/block -> 12672 blocks (exact)
  convert_w1_kernel<<<12672, 256, 0, stream>>>(W1, wt);
  scan_kernel<<<NWG, TPB, 0, stream>>>(x, s0, b1i, b1r, W2, b2i, b2r, Wd, bd,
                                       wt, ctr, out);
}
